// TensorTrainGaussian_38860864094824
// MI455X (gfx1250) — matrix operand order repaired and checked by reading, not compiled, not run on hardware
//
#include <hip/hip_runtime.h>
#include <cmath>

typedef __attribute__((ext_vector_type(16))) _Float16 v16h;
typedef __attribute__((ext_vector_type(8)))  float    v8f;
typedef __attribute__((ext_vector_type(4)))  float    f4;
typedef __attribute__((ext_vector_type(2)))  float    f2;
typedef __attribute__((ext_vector_type(2)))  _Float16 h2;
typedef __attribute__((ext_vector_type(4)))  unsigned v4u;
typedef __attribute__((ext_vector_type(8)))  int      v8i;
typedef __attribute__((ext_vector_type(4)))  int      v4i;

#define KD 32
#define MSTEPS 32
#define SPW 4      // sample-chains per wave
#define WAVES 8    // waves per block
#define LOG2E 1.4426950408889634f
#define HALF_LOG_2PI 0.9189385332046727f
#define EPSF 2.220446049250313e-16f
#define SWAPX16 0x401F          // ds_swizzle imm: group-of-32, xor=0x10, and=0x1f
#define A_DWORDS (MSTEPS * 2 * 32 * 16)   // 32768 dwords = 128 KB swizzled a-table

// ---------------- setup kernel 1: logsumexp over axis=1 (rows i) of W_logits[m,:,j]
__global__ void lse_kernel(const float* __restrict__ W, float* __restrict__ lse) {
    int m = blockIdx.x, j = threadIdx.x;
    float mx = -INFINITY;
    for (int i = 0; i < KD; ++i) mx = fmaxf(mx, W[(m * KD + i) * KD + j]);
    float s = 0.f;
    for (int i = 0; i < KD; ++i) s += __expf(W[(m * KD + i) * KD + j] - mx);
    lse[m * KD + j] = mx + __logf(s);
}

// ---------------- setup kernel 2: softmax of wk0_logits (K=32, one wave)
__global__ void wk0_kernel(const float* __restrict__ wl, float* __restrict__ wk0) {
    int j = threadIdx.x;
    float v = wl[j];
    float mx = v;
    for (int d = 16; d >= 1; d >>= 1) mx = fmaxf(mx, __shfl_xor(mx, d, 32));
    float e = __expf(v - mx);
    float s = e;
    for (int d = 16; d >= 1; d >>= 1) s += __shfl_xor(s, d, 32);
    wk0[j] = e / s;
}

// ---------------- setup kernel 3: build swizzled coefficient tables in WMMA f16
// A-matrix (16x32) lane layout: half = lane>>4, row = I*16+(lane&15),
// element e -> col = 16*half + 8*(e&1) + (e>>1) (this kernel's k order; B packing uses the same).
// result[r][c] = exp2( a_m[r][c] + b[r][c]*x + c[r][c]*x^2 )
__global__ void tables_kernel(const float* __restrict__ W, const float* __restrict__ mu,
                              const float* __restrict__ ps, const float* __restrict__ lse,
                              float* __restrict__ aSw, float* __restrict__ bSw,
                              float* __restrict__ cSw) {
    int m = blockIdx.x >> 1, I = blockIdx.x & 1, lane = threadIdx.x;
    int half = lane >> 4, row = I * 16 + (lane & 15);
    for (int e = 0; e < 16; ++e) {
        int col = 16 * half + 8 * (e & 1) + (e >> 1);
        float mv = mu[row * KD + col];
        float pv = ps[row * KD + col];
        float sg = (pv > 0.f) ? (pv + log1pf(__expf(-pv))) : log1pf(__expf(pv)); // softplus
        float inv = 1.f / (sg * sg);
        float lw  = W[(m * KD + row) * KD + col] - lse[m * KD + col];            // log_softmax
        float a   = (lw - __logf(sg) - HALF_LOG_2PI - 0.5f * mv * mv * inv) * LOG2E;
        aSw[((m * 2 + I) * 32 + lane) * 16 + e] = a;
        if (m == 0) {
            bSw[(I * 32 + lane) * 16 + e] = mv * inv * LOG2E;
            cSw[(I * 32 + lane) * 16 + e] = -0.5f * inv * LOG2E;
        }
    }
}

__device__ __forceinline__ unsigned pack_f16x2(float lo, float hi) {
    h2 v;
    v.x = (_Float16)lo;
    v.y = (_Float16)hi;
    return __builtin_bit_cast(unsigned, v);
}

union U16 {
    unsigned u[8];
    v16h h;
};

// load 16 floats (f4-aligned) into 8 float2 pairs (for v_pk_fma_f32 formation)
__device__ __forceinline__ void load16_f2(const float* p, f2* d) {
#pragma unroll
    for (int v = 0; v < 4; ++v) {
        f4 t = ((const f4*)p)[v];
        d[2 * v + 0].x = t.x; d[2 * v + 0].y = t.y;
        d[2 * v + 1].x = t.z; d[2 * v + 1].y = t.w;
    }
}

// ---------------- main kernel: 8 waves/block, SPW sample-chains per wave
__global__ __launch_bounds__(256) void tt_chain_kernel(
    const float* __restrict__ X, const float* __restrict__ aSw,
    const float* __restrict__ bSw, const float* __restrict__ cSw,
    const float* __restrict__ wk0, float* __restrict__ out) {
    extern __shared__ float sA[];   // 128 KB: whole swizzled a-table, block-shared

    const int lane = threadIdx.x & 31;
    const int wave = threadIdx.x >> 5;
    const int n0   = (blockIdx.x * WAVES + wave) * SPW;
    const int half = lane >> 4;
    // byte-permute selector for the packed half-wave swap:
    // low half lanes need [p.lo | sw.lo], high half lanes need [sw.hi | p.hi]
    const unsigned sel = (half == 0) ? 0x01000504u : 0x07060302u;

    // ---- stage the 128 KB a-table into LDS once per block via the TDM ----
#if __has_builtin(__builtin_amdgcn_tensor_load_to_lds) && __has_builtin(__builtin_amdgcn_s_wait_tensorcnt)
    if (wave == 0) {
        unsigned long long ga = (unsigned long long)(const void*)aSw;
        unsigned lds = (unsigned)(unsigned long long)(void*)sA;
        v4u g0;
        g0[0] = 1u;                                   // count=1, user-mode descriptor
        g0[1] = lds;                                  // lds_addr (bytes)
        g0[2] = (unsigned)ga;                         // global_addr[31:0]
        g0[3] = ((unsigned)(ga >> 32) & 0x01FFFFFFu)  // global_addr[56:32]
                | 0x80000000u;                        // type=2 ("image") in bits[127:126]
        v8i g1 = {};
        g1[0] = 0x20000;                              // data_size=2 -> 4-byte elements
        g1[1] = (int)((A_DWORDS & 0xFFFFu) << 16);    // tensor_dim0[15:0] -> bits[63:48]
        g1[2] = (int)((A_DWORDS >> 16) & 0xFFFFu)     // tensor_dim0[31:16] -> bits[79:64]
                | (1 << 16);                          // tensor_dim1 = 1
        g1[3] = (int)((A_DWORDS & 0xFFFFu) << 16);    // tile_dim0 -> bits[127:112]
        g1[4] = 1;                                    // tile_dim1 = 1
        g1[5] = A_DWORDS;                             // tensor_dim0_stride
        v4i g2 = {}, g3 = {};
        v8i g4 = {};
        __builtin_amdgcn_tensor_load_to_lds(g0, g1, g2, g3, g4, 0);
        __builtin_amdgcn_s_wait_tensorcnt(0);
    }
    __syncthreads();
#else
    {
        const f4* src = (const f4*)aSw;
        f4* dst = (f4*)sA;
        for (int i = threadIdx.x; i < A_DWORDS / 4; i += WAVES * 32) dst[i] = src[i];
        __syncthreads();
    }
#endif

    // persistent per-lane quadratic coefficients as float2 pairs
    f2 b0[8], b1[8], c0[8], c1[8];
    load16_f2(bSw + (0 * 32 + lane) * 16, b0);
    load16_f2(bSw + (1 * 32 + lane) * 16, b1);
    load16_f2(cSw + (0 * 32 + lane) * 16, c0);
    load16_f2(cSw + (1 * 32 + lane) * 16, c1);

    // X rows: lane L holds X[n][L]
    float xr[SPW];
#pragma unroll
    for (int s = 0; s < SPW; ++s) xr[s] = X[(n0 + s) * MSTEPS + lane];

    // P = prod^T state: tiles [I*2+J] of 16x16 f32 per sample; init = identity
    v8f tP[SPW][4];
#pragma unroll
    for (int s = 0; s < SPW; ++s)
#pragma unroll
        for (int I = 0; I < 2; ++I)
#pragma unroll
            for (int J = 0; J < 2; ++J)
#pragma unroll
                for (int r = 0; r < 8; ++r)
                    tP[s][I * 2 + J][r] = (I == J && (r + 8 * half) == (lane & 15)) ? 1.f : 0.f;

#pragma unroll 1
    for (int m = 0; m < MSTEPS; ++m) {
        // per-step affine coefficients from LDS (shared by all SPW samples)
        f2 a0[8], a1[8];
        load16_f2(sA + ((m * 2 + 0) * 32 + lane) * 16, a0);
        load16_f2(sA + ((m * 2 + 1) * 32 + lane) * 16, a1);

#pragma unroll
        for (int s = 0; s < SPW; ++s) {
            // uniform broadcast of X[n][m] via v_readlane (no LDS op)
            float x = __builtin_bit_cast(float,
                        __builtin_amdgcn_readlane(__builtin_bit_cast(int, xr[s]), m));
            f2 xv; xv.x = x; xv.y = x;

            // A fragments: packed quadratic (v_pk_fma_f32) then exp2 per element
            v16h A0, A1;
#pragma unroll
            for (int e = 0; e < 8; ++e) {
                f2 t0 = __builtin_elementwise_fma(
                            __builtin_elementwise_fma(c0[e], xv, b0[e]), xv, a0[e]);
                f2 t1 = __builtin_elementwise_fma(
                            __builtin_elementwise_fma(c1[e], xv, b1[e]), xv, a1[e]);
                A0[2 * e + 0] = (_Float16)__builtin_amdgcn_exp2f(t0.x);
                A0[2 * e + 1] = (_Float16)__builtin_amdgcn_exp2f(t0.y);
                A1[2 * e + 0] = (_Float16)__builtin_amdgcn_exp2f(t1.x);
                A1[2 * e + 1] = (_Float16)__builtin_amdgcn_exp2f(t1.y);
            }

            // B fragments from carried state: pack f16x2, ds_swizzle SWAPX16, v_perm
            U16 Bu0, Bu1;
#pragma unroll
            for (int J = 0; J < 2; ++J) {
                U16* B = J ? &Bu1 : &Bu0;
#pragma unroll
                for (int r = 0; r < 8; ++r) {
                    unsigned p  = pack_f16x2(tP[s][J][r], tP[s][2 + J][r]);
                    unsigned sw = (unsigned)__builtin_amdgcn_ds_swizzle((int)p, SWAPX16);
                    B->u[r] = __builtin_amdgcn_perm(p, sw, sel);
                }
            }

            // P_new tile(I,J) = result[I-rows,:] x P[:,J-cols], full K=32 in one WMMA
            v8f z = {};
            tP[s][0] = __builtin_amdgcn_wmma_f32_16x16x32_f16(false, A0, false, Bu0.h, (short)0, z, false, false);
            tP[s][1] = __builtin_amdgcn_wmma_f32_16x16x32_f16(false, A0, false, Bu1.h, (short)0, z, false, false);
            tP[s][2] = __builtin_amdgcn_wmma_f32_16x16x32_f16(false, A1, false, Bu0.h, (short)0, z, false, false);
            tP[s][3] = __builtin_amdgcn_wmma_f32_16x16x32_f16(false, A1, false, Bu1.h, (short)0, z, false, false);
        }
    }

    // likelihood = sum_k wk0[k] * colsum_k(P)  (colsum of P == rowsum of prod)
#pragma unroll
    for (int s = 0; s < SPW; ++s) {
        float cs0 = 0.f, cs1 = 0.f;
#pragma unroll
        for (int r = 0; r < 8; ++r) {
            cs0 += tP[s][0][r] + tP[s][2][r];   // J=0 tiles: cols 0..15
            cs1 += tP[s][1][r] + tP[s][3][r];   // J=1 tiles: cols 16..31
        }
        cs0 += __shfl_xor(cs0, 16, 32);
        cs1 += __shfl_xor(cs1, 16, 32);
        float val = cs0 * wk0[lane & 15] + cs1 * wk0[16 + (lane & 15)];
        val += __shfl_xor(val, 8, 32);
        val += __shfl_xor(val, 4, 32);
        val += __shfl_xor(val, 2, 32);
        val += __shfl_xor(val, 1, 32);
        if (lane == 0) out[n0 + s] = __logf(val + EPSF);
    }
}

extern "C" void kernel_launch(void* const* d_in, const int* in_sizes, int n_in,
                              void* d_out, int out_size, void* d_ws, size_t ws_size,
                              hipStream_t stream) {
    const float* X          = (const float*)d_in[0];
    const float* wk0_logits = (const float*)d_in[1];
    const float* W_logits   = (const float*)d_in[2];
    const float* mu         = (const float*)d_in[3];
    const float* pre_sigma  = (const float*)d_in[4];
    float* out = (float*)d_out;

    char* ws = (char*)d_ws;
    float* aSw = (float*)(ws);            // M*2*32*16 f32 = 131072 B
    float* bSw = (float*)(ws + 131072);   // 4096 B
    float* cSw = (float*)(ws + 135168);   // 4096 B
    float* wk0 = (float*)(ws + 139264);   // 128 B
    float* lse = (float*)(ws + 139392);   // 4096 B

    lse_kernel<<<MSTEPS, KD, 0, stream>>>(W_logits, lse);
    wk0_kernel<<<1, KD, 0, stream>>>(wk0_logits, wk0);
    tables_kernel<<<MSTEPS * 2, 32, 0, stream>>>(W_logits, mu, pre_sigma, lse, aSw, bSw, cSw);

    int N = in_sizes[0] / MSTEPS;         // 65536
    tt_chain_kernel<<<N / (WAVES * SPW), 256, A_DWORDS * 4, stream>>>(X, aSw, bSw, cSw, wk0, out);
}
